// LSTM_RNN_Model_32521492365624
// MI455X (gfx1250) — matrix operand order repaired and checked by reading, not compiled, not run on hardware
//
#include <hip/hip_runtime.h>

// ---------------------------------------------------------------------------
// 2-layer LSTM, B=256, T=512, H=1024, future=0.
// Strategy: bf16 WMMA (v_wmma_f32_16x16x32_bf16) with fp32 accumulation.
// Weights pre-transposed to bf16 [K=1024][N=4096] in workspace -> L2-resident
// (24MB << 192MB L2), each weight element read once per M-block (4x) per timestep.
// Per timestep: 2 fused layer kernels + 1 tiny output-projection kernel.
// ---------------------------------------------------------------------------

#define HDIM  1024
#define FOURH 4096
#define BDIM  256
#define TDIM  512

typedef __attribute__((ext_vector_type(16))) __bf16 v16bf;
typedef __attribute__((ext_vector_type(8)))  float  v8f;

union Frag {
    uint4 q[2];
    v16bf v;
};

__device__ __forceinline__ unsigned short f2bf(float f) {
    union { float f; unsigned u; } v;
    v.f = f;
    unsigned u = v.u;
    unsigned r = u + 0x7fffu + ((u >> 16) & 1u);     // round-to-nearest-even
    if ((u & 0x7f800000u) == 0x7f800000u) r = u;     // inf/nan: truncate
    return (unsigned short)(r >> 16);
}

__device__ __forceinline__ float sigf(float v) {
    return 1.0f / (1.0f + __expf(-v));
}
__device__ __forceinline__ float tanh_fast(float v) {
    return 2.0f / (1.0f + __expf(-2.0f * v)) - 1.0f;
}

// ---------------------------------------------------------------------------
// Prep kernels (run once per launch)
// ---------------------------------------------------------------------------

// src: fp32 [4096][1024] (W rows n, cols k). dst: bf16 [1024][4096] (Wt[k][n]).
__global__ __launch_bounds__(256) void k_transpose_bf16(
        const float* __restrict__ src, unsigned short* __restrict__ dst) {
    int flat = blockIdx.x * 256 + threadIdx.x;   // index into dst, n fastest
    int n = flat & (FOURH - 1);
    int k = flat >> 12;
    dst[flat] = f2bf(src[n * HDIM + k]);
}

__global__ __launch_bounds__(256) void k_bias(
        const float* __restrict__ a, const float* __restrict__ b,
        const float* __restrict__ c, const float* __restrict__ d,
        float* __restrict__ o1, float* __restrict__ o2) {
    int i = blockIdx.x * 256 + threadIdx.x;
    if (i < FOURH) o1[i] = a[i] + b[i];
    else { int j = i - FOURH; o2[j] = c[j] + d[j]; }
}

__global__ __launch_bounds__(256) void k_zero(uint4* __restrict__ p) {
    uint4 z; z.x = 0u; z.y = 0u; z.z = 0u; z.w = 0u;
    p[blockIdx.x * 256 + threadIdx.x] = z;
}

// ---------------------------------------------------------------------------
// Fused LSTM layer step.
//   gates[b, g*H + j] = sum_k A0[b,k]*W0[k, g*H+j] (+ A1@W1 if NG==2)
//                       + bias[g*H+j] (+ x[b,t]*wih[g*H+j] if NG==1)
//   then cell update -> c (in place, fp32), h (bf16 [+ fp32 for layer 2]).
// Grid: (256/64, 1024/64) = (4,16). Block: 256 threads = 8 waves.
// Wave w owns N-tiles {2w, 2w+1} (nt = gate*4 + jsub) for all 4 M-tiles:
// weight fragments are loaded exactly once per block (no redundancy).
// Per K-step: batch 8 ds_load_b128 (all 4 M-tile A frags) in one clause ->
// single dscnt wait -> 8 back-to-back WMMAs (XDL co-execution friendly).
// ---------------------------------------------------------------------------
template <int NG>
__global__ __launch_bounds__(256) void lstm_layer(
        const unsigned short* __restrict__ A0,   // h input 0, bf16 [256][1024]
        const unsigned short* __restrict__ W0,   // bf16 [1024][4096]
        const unsigned short* __restrict__ A1,   // h input 1 (NG==2)
        const unsigned short* __restrict__ W1,   // bf16 [1024][4096] (NG==2)
        const float* __restrict__ bias,          // [4096]
        const float* __restrict__ x,             // [B][T] (NG==1)
        int t,
        const float* __restrict__ wih,           // [4096] (NG==1)
        float* __restrict__ cst,                 // c state fp32 [256][1024]
        unsigned short* __restrict__ hb,         // h out bf16 [256][1024]
        float* __restrict__ hf)                  // h out fp32 (NG==2)
{
    extern __shared__ char smem[];               // 64 KB
    uint4* sQ = (uint4*)smem;                    // A chunk, [64 rows][64 uint4]
    const unsigned short* sH = (const unsigned short*)smem;
    float* gsm = (float*)smem;                   // gate staging [64][256]

    const int tid  = threadIdx.x;
    const int wave = tid >> 5;
    const int lane = tid & 31;
    const int m0 = blockIdx.x * 64;
    const int j0 = blockIdx.y * 64;

    const int nt0  = wave * 2;
    const int nt1  = nt0 + 1;
    const int col0 = (nt0 >> 2) * HDIM + j0 + (nt0 & 3) * 16;
    const int col1 = (nt1 >> 2) * HDIM + j0 + (nt1 & 3) * 16;

    const int lm = lane & 15;    // M within tile (A), N within tile (C/D)
    const int lh = lane >> 4;    // lane half

    v8f acc[4][2];
    v8f vz = {0.f, 0.f, 0.f, 0.f, 0.f, 0.f, 0.f, 0.f};
#pragma unroll
    for (int mt = 0; mt < 4; ++mt) { acc[mt][0] = vz; acc[mt][1] = vz; }

    for (int g = 0; g < NG; ++g) {
        const unsigned short* A = (g == 0) ? A0 : A1;
        const unsigned short* W = (g == 0) ? W0 : W1;
        for (int kc = 0; kc < HDIM; kc += 512) {
            __syncthreads();
            // Cooperative stage of A chunk: rows m0..m0+63, K kc..kc+511.
            // 64 rows * 64 uint4/row = 4096 uint4, 16 per thread.
#pragma unroll
            for (int i = 0; i < 16; ++i) {
                int fl = tid + i * 256;
                int r = fl >> 6;
                int q = fl & 63;
                sQ[fl] = *(const uint4*)(A + (size_t)(m0 + r) * HDIM + kc + q * 8);
            }
            __syncthreads();
            for (int ks = 0; ks < 16; ++ks) {
                const int kb = ks * 32;
                // B fragments: lane <-> N column lm, elements over K in table order (same as A).
                Frag fb0, fb1;
#pragma unroll
                for (int s = 0; s < 16; ++s) {
                    const int kk = kc + kb + (s < 8 ? 8 * lh + s : 16 + 8 * lh + (s - 8));
                    const unsigned short* wrow = W + (size_t)kk * FOURH;
                    ((unsigned short*)&fb0)[s] = wrow[col0 + lm];
                    ((unsigned short*)&fb1)[s] = wrow[col1 + lm];
                }
                // A fragments (16x32 bf16): elements 0..7 = K kb+8*lh..,
                // elements 8..15 = K kb+16+8*lh.. ; 8 b128 LDS loads batched.
                Frag fa[4];
#pragma unroll
                for (int mt = 0; mt < 4; ++mt) {
                    const unsigned short* ar = sH + (mt * 16 + lm) * 512;
                    fa[mt].q[0] = *(const uint4*)(ar + kb + 8 * lh);
                    fa[mt].q[1] = *(const uint4*)(ar + kb + 16 + 8 * lh);
                }
#pragma unroll
                for (int mt = 0; mt < 4; ++mt) {
                    acc[mt][0] = __builtin_amdgcn_wmma_f32_16x16x32_bf16(
                        false, fa[mt].v, false, fb0.v, (short)0, acc[mt][0],
                        false, false);
                    acc[mt][1] = __builtin_amdgcn_wmma_f32_16x16x32_bf16(
                        false, fa[mt].v, false, fb1.v, (short)0, acc[mt][1],
                        false, false);
                }
            }
        }
    }

    __syncthreads();
    // Stage accumulators: C/D layout lane -> N=lm, VGPR r -> M = 8*lh + r.
#pragma unroll
    for (int mt = 0; mt < 4; ++mt) {
#pragma unroll
        for (int p = 0; p < 2; ++p) {
            const int nt = nt0 + p;
            v8f a = acc[mt][p];
#pragma unroll
            for (int r = 0; r < 8; ++r) {
                gsm[(mt * 16 + lh * 8 + r) * 256 + nt * 16 + lm] = a[r];
            }
        }
    }
    __syncthreads();

    // Fused LSTM cell update: 64x64 elements, 16 per thread.
    for (int e = 0; e < 16; ++e) {
        const int idx = tid + e * 256;
        const int m  = idx >> 6;
        const int jj = idx & 63;
        const int b  = m0 + m;
        const int j  = j0 + jj;

        float gi = gsm[m * 256 + 0 * 64 + jj] + bias[0 * HDIM + j];
        float gf = gsm[m * 256 + 1 * 64 + jj] + bias[1 * HDIM + j];
        float gg = gsm[m * 256 + 2 * 64 + jj] + bias[2 * HDIM + j];
        float go = gsm[m * 256 + 3 * 64 + jj] + bias[3 * HDIM + j];
        if (NG == 1) {
            const float xv = x[b * TDIM + t];
            gi += xv * wih[0 * HDIM + j];
            gf += xv * wih[1 * HDIM + j];
            gg += xv * wih[2 * HDIM + j];
            go += xv * wih[3 * HDIM + j];
        }
        const float co = cst[(size_t)b * HDIM + j];
        const float cn = sigf(gf) * co + sigf(gi) * tanh_fast(gg);
        cst[(size_t)b * HDIM + j] = cn;
        const float hn = sigf(go) * tanh_fast(cn);
        hb[(size_t)b * HDIM + j] = f2bf(hn);
        if (NG == 2) hf[(size_t)b * HDIM + j] = hn;
    }
}

// ---------------------------------------------------------------------------
// out[b, t] = dot(h2f[b, :], W_out) + b_out.  Deterministic tree reduction.
// Grid: 32 blocks x 256 threads; wave w of block blk handles row blk*8 + w.
// ---------------------------------------------------------------------------
__global__ __launch_bounds__(256) void k_outdot(
        const float* __restrict__ h2f, const float* __restrict__ wout,
        const float* __restrict__ bout, float* __restrict__ out, int t) {
    const int wave = threadIdx.x >> 5;
    const int lane = threadIdx.x & 31;
    const int b = blockIdx.x * 8 + wave;
    float s = 0.0f;
    for (int i = lane; i < HDIM; i += 32)
        s += h2f[(size_t)b * HDIM + i] * wout[i];
#pragma unroll
    for (int off = 16; off > 0; off >>= 1)
        s += __shfl_down(s, off, 32);
    if (lane == 0) out[(size_t)b * TDIM + t] = s + bout[0];
}

// ---------------------------------------------------------------------------
// Host side
// ---------------------------------------------------------------------------
extern "C" void kernel_launch(void* const* d_in, const int* in_sizes, int n_in,
                              void* d_out, int out_size, void* d_ws, size_t ws_size,
                              hipStream_t stream) {
    (void)in_sizes; (void)n_in; (void)out_size; (void)ws_size;

    const float* x     = (const float*)d_in[0];   // [256][512]
    const float* Wih1  = (const float*)d_in[1];   // [4096][1]
    const float* Whh1  = (const float*)d_in[2];   // [4096][1024]
    const float* bih1  = (const float*)d_in[3];
    const float* bhh1  = (const float*)d_in[4];
    const float* Wih2  = (const float*)d_in[5];   // [4096][1024]
    const float* Whh2  = (const float*)d_in[6];   // [4096][1024]
    const float* bih2  = (const float*)d_in[7];
    const float* bhh2  = (const float*)d_in[8];
    const float* Wout  = (const float*)d_in[9];   // [1][1024]
    const float* bout  = (const float*)d_in[10];
    float* out = (float*)d_out;                   // [256][512]

    const size_t WT_BYTES = (size_t)HDIM * FOURH * 2;  // 8 MB each
    char* ws = (char*)d_ws;
    unsigned short* Wt1 = (unsigned short*)(ws);                 // W_hh1^T bf16
    unsigned short* Wt2 = (unsigned short*)(ws + WT_BYTES);      // W_ih2^T bf16
    unsigned short* Wt3 = (unsigned short*)(ws + 2 * WT_BYTES);  // W_hh2^T bf16
    float* b1 = (float*)(ws + 3 * WT_BYTES);
    float* b2 = b1 + FOURH;
    char* st = ws + 3 * WT_BYTES + 32768;
    const size_t HB_BYTES = (size_t)BDIM * HDIM * 2;   // 512 KB
    const size_t HF_BYTES = (size_t)BDIM * HDIM * 4;   // 1 MB
    unsigned short* h1b[2] = {(unsigned short*)st,
                              (unsigned short*)(st + HB_BYTES)};
    unsigned short* h2b[2] = {(unsigned short*)(st + 2 * HB_BYTES),
                              (unsigned short*)(st + 3 * HB_BYTES)};
    float* c1  = (float*)(st + 4 * HB_BYTES);
    float* c2  = (float*)(st + 4 * HB_BYTES + HF_BYTES);
    float* h2f = (float*)(st + 4 * HB_BYTES + 2 * HF_BYTES);
    const size_t state_bytes = 4 * HB_BYTES + 3 * HF_BYTES;      // 5 MB

    // --- one-time prep ---
    const int tgrid = (HDIM * FOURH) / 256;                      // 16384
    k_transpose_bf16<<<tgrid, 256, 0, stream>>>(Whh1, Wt1);
    k_transpose_bf16<<<tgrid, 256, 0, stream>>>(Wih2, Wt2);
    k_transpose_bf16<<<tgrid, 256, 0, stream>>>(Whh2, Wt3);
    k_bias<<<(2 * FOURH) / 256, 256, 0, stream>>>(bih1, bhh1, bih2, bhh2, b1, b2);
    k_zero<<<(int)(state_bytes / 16 / 256), 256, 0, stream>>>((uint4*)st);

    // --- scan over timesteps ---
    const dim3 lgrid(BDIM / 64, HDIM / 64);   // (4,16)
    const size_t smem = 64 * 1024;
    for (int t = 0; t < TDIM; ++t) {
        const int rd = t & 1, wr = rd ^ 1;
        lstm_layer<1><<<lgrid, 256, smem, stream>>>(
            h1b[rd], Wt1, nullptr, nullptr, b1, x, t, Wih1,
            c1, h1b[wr], nullptr);
        lstm_layer<2><<<lgrid, 256, smem, stream>>>(
            h1b[wr], Wt2, h2b[rd], Wt3, b2, nullptr, t, nullptr,
            c2, h2b[wr], h2f);
        k_outdot<<<BDIM / 8, 256, 0, stream>>>(h2f, Wout, bout, out, t);
    }
}
